// Block_14130442404312
// MI455X (gfx1250) — matrix operand order repaired and checked by reading, not compiled, not run on hardware
//
#include <hip/hip_runtime.h>
#include <hip/hip_bf16.h>
#include <math.h>

// ---------------------------------------------------------------------------
// Transformer block (self-attn + cross-attn + MLP) for gfx1250 (MI455X).
// - All GEMMs / attention einsums: v_wmma_f32_16x16x32_bf16 (wave32).
// - Global->LDS staging: Tensor Data Mover (tensor_load_to_lds), double
//   buffered, synchronized with s_wait_tensorcnt + workgroup barriers.
// - Weights pre-transposed to [N,K] so every WMMA fragment is two contiguous
//   ds_load_b128; V stored [B,H,D,L] so attention V tiles stage row-major.
// ---------------------------------------------------------------------------

typedef __bf16 bf16;
typedef __attribute__((ext_vector_type(16))) __bf16 v16bf;
typedef __attribute__((ext_vector_type(8)))  __bf16 v8bf;
typedef __attribute__((ext_vector_type(8)))  float  v8f;
typedef unsigned int uint32x4 __attribute__((ext_vector_type(4)));
typedef int          int32x4  __attribute__((ext_vector_type(4)));
typedef int          int32x8  __attribute__((ext_vector_type(8)));

#define B_  4
#define T_  1024
#define TP_ 256
#define C_  1024
#define H_  16
#define D_  64

#define NEGV (-3.0e38f)
#define NEGT (-1.0e37f)

static __device__ __forceinline__ bf16 tobf(float f) {
  unsigned u = __builtin_bit_cast(unsigned, f);
  unsigned r = u + 0x7fffu + ((u >> 16) & 1u);  // round-to-nearest-even
  return __builtin_bit_cast(bf16, (unsigned short)(r >> 16));
}

static __device__ __forceinline__ v16bf cat16(v8bf lo, v8bf hi) {
  union { v16bf v; v8bf h[2]; } u;
  u.h[0] = lo; u.h[1] = hi;
  return u.v;
}

static __device__ __forceinline__ v8f vzero8() {
  v8f z;
#pragma unroll
  for (int i = 0; i < 8; i++) z[i] = 0.0f;
  return z;
}

// ---------------------------------------------------------------------------
// Tensor Data Mover: 2-D tile load, global -> LDS, with LDS row padding.
// D# layout per CDNA5 ISA ch. 8.3/8.4 (group0 128b, group1 256b).
//   tile: tile1 rows x tile0 elements (bf16), row stride = stride0 elements.
//   LDS: rows padded by pad_amount DWORDs every (2 << pad_int_code) DWORDs.
// ---------------------------------------------------------------------------
static __device__ __forceinline__ void tdm_load_2d(
    unsigned lds_addr, const void* gptr, unsigned tile0, unsigned tile1,
    unsigned long long stride0, unsigned pad_int_code, unsigned pad_amt_code) {
  unsigned long long ga = (unsigned long long)(uintptr_t)gptr;
  uint32x4 g0;
  g0[0] = 1u;                                    // count=1, user descriptor
  g0[1] = lds_addr;                              // LDS byte address
  g0[2] = (unsigned)ga;                          // global_addr[31:0]
  g0[3] = (unsigned)((ga >> 32) & 0x01FFFFFFu)   // global_addr[56:32]
        | (2u << 30);                            // type = 2 ("image")
  int32x8 g1;
  g1[0] = (int)((1u << 16)                       // data_size = 2 bytes
        | (1u << 20)                             // pad_enable
        | (pad_int_code << 22)                   // pad every 2<<code DWORDs
        | (pad_amt_code << 25));                 // pad (code+1) DWORDs
  g1[1] = (int)((tile0 & 0xFFFFu) << 16);        // tensor_dim0[15:0]
  g1[2] = (int)((tile0 >> 16) | ((tile1 & 0xFFFFu) << 16));  // dim0 hi|dim1 lo
  g1[3] = (int)((tile1 >> 16) | ((tile0 & 0xFFFFu) << 16));  // dim1 hi|tile_dim0
  g1[4] = (int)(tile1 & 0xFFFFu);                // tile_dim1 (tile_dim2 = 0)
  g1[5] = (int)(unsigned)stride0;                // tensor_dim0_stride[31:0]
  g1[6] = (int)(unsigned)((stride0 >> 32) & 0xFFFFu);
  g1[7] = 0;
  int32x4 z4 = {0, 0, 0, 0};
#if defined(__clang_major__) && (__clang_major__ >= 23)
  int32x8 z8 = {0, 0, 0, 0, 0, 0, 0, 0};
  __builtin_amdgcn_tensor_load_to_lds(g0, g1, z4, z4, z8, 0);
#else
  __builtin_amdgcn_tensor_load_to_lds(g0, g1, z4, z4, 0);
#endif
}

static __device__ __forceinline__ unsigned lds_addr_of(const void* p) {
  return (unsigned)(uintptr_t)p;  // low 32 bits of flat shared address
}

// ---------------------------------------------------------------------------
// Weight convert + transpose: fp32 [K,N] -> bf16 [N,K] (LDS tile transpose)
// ---------------------------------------------------------------------------
__global__ __launch_bounds__(256) void cvt_t_kernel(
    const float* __restrict__ in, bf16* __restrict__ out, int K, int N) {
  __shared__ float tile[32][33];
  const int tx = threadIdx.x & 31, ty = threadIdx.x >> 5;  // 32x8 threads
  const int n0 = blockIdx.x * 32, k0 = blockIdx.y * 32;
#pragma unroll
  for (int r = 0; r < 32; r += 8)
    tile[ty + r][tx] = in[(size_t)(k0 + ty + r) * N + n0 + tx];
  __syncthreads();
#pragma unroll
  for (int r = 0; r < 32; r += 8)
    out[(size_t)(n0 + ty + r) * K + k0 + tx] = tobf(tile[tx][ty + r]);
}

// ---------------------------------------------------------------------------
// LayerNorm over C=1024, one block (8 waves) per token, bf16 output
// ---------------------------------------------------------------------------
__global__ __launch_bounds__(256) void ln_kernel(
    const float* __restrict__ x, const float* __restrict__ w,
    const float* __restrict__ b, bf16* __restrict__ out) {
  const int tok = blockIdx.x;
  const float* xr = x + (size_t)tok * C_;
  float v[4], s = 0.f, s2 = 0.f;
#pragma unroll
  for (int k = 0; k < 4; k++) {
    float t = xr[threadIdx.x + 256 * k];
    v[k] = t; s += t; s2 += t * t;
  }
#pragma unroll
  for (int off = 16; off >= 1; off >>= 1) {
    s  += __shfl_xor(s,  off, 32);
    s2 += __shfl_xor(s2, off, 32);
  }
  __shared__ float rs[8], rs2[8];
  const int wave = threadIdx.x >> 5, lane = threadIdx.x & 31;
  if (lane == 0) { rs[wave] = s; rs2[wave] = s2; }
  __syncthreads();
  float S = 0.f, S2 = 0.f;
#pragma unroll
  for (int i = 0; i < 8; i++) { S += rs[i]; S2 += rs2[i]; }
  const float mean = S * (1.0f / C_);
  const float var  = S2 * (1.0f / C_) - mean * mean;
  const float inv  = rsqrtf(var + 1e-5f);
  bf16* o = out + (size_t)tok * C_;
#pragma unroll
  for (int k = 0; k < 4; k++) {
    int c = threadIdx.x + 256 * k;
    o[c] = tobf((v[k] - mean) * inv * w[c] + b[c]);
  }
}

// ---------------------------------------------------------------------------
// pocket: fp32 -> bf16 + per-row (sum|x| == 0) mask byte
// ---------------------------------------------------------------------------
__global__ __launch_bounds__(256) void pocket_kernel(
    const float* __restrict__ p, bf16* __restrict__ pbf,
    unsigned char* __restrict__ pmask) {
  const int row = blockIdx.x;  // b*Tp + tp
  const float* xr = p + (size_t)row * C_;
  bf16* o = pbf + (size_t)row * C_;
  float s = 0.f;
#pragma unroll
  for (int k = 0; k < 4; k++) {
    float t = xr[threadIdx.x + 256 * k];
    s += fabsf(t);
    o[threadIdx.x + 256 * k] = tobf(t);
  }
#pragma unroll
  for (int off = 16; off >= 1; off >>= 1) s += __shfl_xor(s, off, 32);
  __shared__ float rs[8];
  const int wave = threadIdx.x >> 5, lane = threadIdx.x & 31;
  if (lane == 0) rs[wave] = s;
  __syncthreads();
  if (threadIdx.x == 0) {
    float S = 0.f;
    for (int i = 0; i < 8; i++) S += rs[i];
    pmask[row] = (S == 0.0f) ? 1 : 0;
  }
}

// ---------------------------------------------------------------------------
// GEMM: Y[M,N] = A[M,K](bf16) x WT[N,K](bf16)^T + bias, WMMA bf16.
// Block tile 128x128, BK=32, 8 waves -> 64x32 each (4x2 accumulators).
// Both LDS tiles are 128 rows x 32 K (stride AS=40, via TDM pad), staged by
// wave 0 with tensor_load_to_lds, double buffered on TENSORcnt.
// mode 0: heads scatter -> bf16 [B,H,L,D] (V chunk as [B,H,D,L])
// mode 1: fp32 residual accumulate (N==1024), optional padding-mask zeroing
// mode 2: exact-erf GELU -> bf16 [M,N]
// ---------------------------------------------------------------------------
#define AS 40  // LDS row stride: 32 elems + 4 DWORD TDM pad

__global__ __launch_bounds__(256) void gemm_bf16_wmma(
    const bf16* __restrict__ A, const bf16* __restrict__ WT,
    const float* __restrict__ bias, void* __restrict__ out,
    const unsigned char* __restrict__ mask,
    int M, int N, int K, int mode, int tbits, int region, int vwhich) {
  __shared__ __align__(16) bf16 Ash[2][128 * AS];
  __shared__ __align__(16) bf16 Bsh[2][128 * AS];
  const int tid  = threadIdx.x;
  const int wave = tid >> 5, lane = tid & 31;
  const int lrow = lane & 15, hi = lane >> 4;
  const int wm = wave >> 2, wn = wave & 3;
  const int m0 = blockIdx.y * 128, n0 = blockIdx.x * 128;

  v8f acc[4][2];
#pragma unroll
  for (int i = 0; i < 4; i++)
#pragma unroll
    for (int j = 0; j < 2; j++) acc[i][j] = vzero8();

  const int nsteps = K >> 5;
  if (wave == 0) {  // prologue: stage K-step 0 into buffer 0
    tdm_load_2d(lds_addr_of(&Ash[0][0]), A + (size_t)m0 * K, 32, 128,
                (unsigned long long)K, 3u, 3u);   // 16-DW rows, 4-DW pad
    tdm_load_2d(lds_addr_of(&Bsh[0][0]), WT + (size_t)n0 * K, 32, 128,
                (unsigned long long)K, 3u, 3u);
  }

  for (int step = 0; step < nsteps; step++) {
    const int buf = step & 1;
    if (wave == 0) {
      if (step + 1 < nsteps) {
        const int bk = (step + 1) << 5;
        tdm_load_2d(lds_addr_of(&Ash[buf ^ 1][0]), A + (size_t)m0 * K + bk,
                    32, 128, (unsigned long long)K, 3u, 3u);
        tdm_load_2d(lds_addr_of(&Bsh[buf ^ 1][0]), WT + (size_t)n0 * K + bk,
                    32, 128, (unsigned long long)K, 3u, 3u);
        __builtin_amdgcn_s_wait_tensorcnt(2);  // current pair done, next in flight
      } else {
        __builtin_amdgcn_s_wait_tensorcnt(0);
      }
    }
    __syncthreads();

    v16bf af[4], bfr[2];
#pragma unroll
    for (int i = 0; i < 4; i++) {
      const int r = wm * 64 + i * 16 + lrow;
      af[i] = cat16(*(const v8bf*)&Ash[buf][r * AS + 8 * hi],
                    *(const v8bf*)&Ash[buf][r * AS + 16 + 8 * hi]);
    }
#pragma unroll
    for (int j = 0; j < 2; j++) {
      const int r = wn * 32 + j * 16 + lrow;  // lane's output column
      bfr[j] = cat16(*(const v8bf*)&Bsh[buf][r * AS + 8 * hi],
                     *(const v8bf*)&Bsh[buf][r * AS + 16 + 8 * hi]);
    }
#pragma unroll
    for (int i = 0; i < 4; i++)
#pragma unroll
      for (int j = 0; j < 2; j++)
        acc[i][j] = __builtin_amdgcn_wmma_f32_16x16x32_bf16(
            false, af[i], false, bfr[j], (short)0, acc[i][j], false, false);
    __syncthreads();  // readers done before buf is DMA-overwritten
  }

  // epilogue
#pragma unroll
  for (int i = 0; i < 4; i++) {
#pragma unroll
    for (int j = 0; j < 2; j++) {
#pragma unroll
      for (int e = 0; e < 8; e++) {
        const int m = m0 + wm * 64 + i * 16 + e + 8 * hi;
        const int n = n0 + wn * 32 + j * 16 + (lane & 15);
        float val = acc[i][j][e] + bias[n];
        if (mode == 1) {  // residual fp32 accumulate, N == 1024
          if (mask && mask[m]) val = 0.0f;
          ((float*)out)[(size_t)m * 1024 + n] += val;
        } else if (mode == 2) {  // exact GELU -> bf16
          float g = 0.5f * val * (1.0f + erff(val * 0.70710678118654752f));
          ((bf16*)out)[(size_t)m * N + n] = tobf(g);
        } else {  // heads scatter
          const int which = n >> 10;
          const int c = n & 1023, h = c >> 6, d = c & 63;
          const int t = m & ((1 << tbits) - 1), b = m >> tbits;
          size_t idx;
          if (which == vwhich)  // V stored [B,H,D,L] for transpose-free tiles
            idx = ((((size_t)(b * H_ + h)) * D_ + d) << tbits) + t;
          else                  // Q/K stored [B,H,L,D]
            idx = (((((size_t)(b * H_ + h)) << tbits) + t)) * D_ + d;
          ((bf16*)out)[(size_t)which * region + idx] = tobf(val);
        }
      }
    }
  }
}

// ---------------------------------------------------------------------------
// Flash attention: block = (q-chunk of 128, head, batch), 8 waves x 16 rows.
// 64-key tiles staged by TDM (K row-major [key][d], V already [B,H,D,L] so
// the V tile is row-major [d][key]); double buffered on TENSORcnt.
// ---------------------------------------------------------------------------
#define KS 72  // LDS row stride: 64 elems + 4 DWORD TDM pad

__global__ __launch_bounds__(256) void attn_wmma(
    const bf16* __restrict__ Q, const bf16* __restrict__ Km,
    const bf16* __restrict__ Vm, const unsigned char* __restrict__ kmask,
    bf16* __restrict__ Y, int Lk, int causal) {
  __shared__ __align__(16) bf16 Ksh[2][64 * KS];      // [key][d]
  __shared__ __align__(16) bf16 VT[2][64 * KS];       // [d][key]
  __shared__ __align__(16) bf16 Psh[8 * 16 * KS];     // per-wave P (16x64)
  const int tid  = threadIdx.x;
  const int wave = tid >> 5, lane = tid & 31;
  const int lrow = lane & 15, hi = lane >> 4;
  const int h = blockIdx.y, b = blockIdx.z;
  const int q0 = blockIdx.x * 128 + wave * 16;
  const bf16* kbase = Km + ((size_t)(b * H_ + h)) * Lk * D_;
  const bf16* vbase = Vm + ((size_t)(b * H_ + h)) * D_ * Lk;

  // Q fragments, D=64 -> two K-steps of 32, resident in registers
  const bf16* qp = Q + (((size_t)(b * H_ + h)) * T_ + q0 + lrow) * D_;
  v16bf qf[2];
#pragma unroll
  for (int kk = 0; kk < 2; kk++)
    qf[kk] = cat16(*(const v8bf*)(qp + kk * 32 + 8 * hi),
                   *(const v8bf*)(qp + kk * 32 + 16 + 8 * hi));

  float rm[8], rl[8];
#pragma unroll
  for (int e = 0; e < 8; e++) { rm[e] = NEGV; rl[e] = 0.f; }
  v8f o[4];
#pragma unroll
  for (int jd = 0; jd < 4; jd++) o[jd] = vzero8();

  const int ntiles = causal ? (blockIdx.x * 2 + 2) : (Lk >> 6);
  if (wave == 0) {  // prologue: stage key-tile 0 (32-DW rows, 4-DW pad)
    tdm_load_2d(lds_addr_of(&Ksh[0][0]), kbase, 64, 64,
                (unsigned long long)D_, 4u, 3u);
    tdm_load_2d(lds_addr_of(&VT[0][0]), vbase, 64, 64,
                (unsigned long long)Lk, 4u, 3u);
  }

  for (int kt = 0; kt < ntiles; kt++) {
    const int buf = kt & 1;
    const int kb = kt * 64;
    if (wave == 0) {
      if (kt + 1 < ntiles) {
        tdm_load_2d(lds_addr_of(&Ksh[buf ^ 1][0]),
                    kbase + (size_t)(kb + 64) * D_, 64, 64,
                    (unsigned long long)D_, 4u, 3u);
        tdm_load_2d(lds_addr_of(&VT[buf ^ 1][0]),
                    vbase + (size_t)(kb + 64), 64, 64,
                    (unsigned long long)Lk, 4u, 3u);
        __builtin_amdgcn_s_wait_tensorcnt(2);
      } else {
        __builtin_amdgcn_s_wait_tensorcnt(0);
      }
    }
    __syncthreads();

    // S = Q K^T over 4 key sub-tiles of 16
    v8f s[4];
#pragma unroll
    for (int j = 0; j < 4; j++) {
      s[j] = vzero8();
#pragma unroll
      for (int kk = 0; kk < 2; kk++) {
        const int r = j * 16 + lrow;  // lane's key column
        v16bf kf = cat16(*(const v8bf*)&Ksh[buf][r * KS + kk * 32 + 8 * hi],
                         *(const v8bf*)&Ksh[buf][r * KS + kk * 32 + 16 + 8 * hi]);
        s[j] = __builtin_amdgcn_wmma_f32_16x16x32_bf16(
            false, qf[kk], false, kf, (short)0, s[j], false, false);
      }
    }

    // scale + masks + online softmax (row m = e + 8*hi, col n = lane&15)
    float sc[8];
#pragma unroll
    for (int e = 0; e < 8; e++) {
      const int qg = q0 + e + 8 * hi;
      float tmax = NEGV;
#pragma unroll
      for (int j = 0; j < 4; j++) {
        const int kg = kb + j * 16 + lrow;
        float vv = s[j][e] * 0.125f;  // 1/sqrt(64)
        if (causal && kg > qg) vv = NEGV;
        if (kmask && kmask[b * Lk + kg]) vv = NEGV;
        s[j][e] = vv;
        tmax = fmaxf(tmax, vv);
      }
#pragma unroll
      for (int off = 8; off >= 1; off >>= 1)
        tmax = fmaxf(tmax, __shfl_xor(tmax, off, 16));
      const float newm = fmaxf(rm[e], tmax);
      const float scale = __expf(rm[e] - newm);  // both-NEGV => 1, but rl==0
      float psum = 0.f;
#pragma unroll
      for (int j = 0; j < 4; j++) {
        float p = (s[j][e] <= NEGT) ? 0.f : __expf(s[j][e] - newm);
        s[j][e] = p;
        psum += p;
      }
#pragma unroll
      for (int off = 8; off >= 1; off >>= 1)
        psum += __shfl_xor(psum, off, 16);
      rl[e] = rl[e] * scale + psum;
      rm[e] = newm;
      sc[e] = scale;
#pragma unroll
      for (int j = 0; j < 4; j++)  // P to per-wave LDS (accum -> A-frag relayout)
        Psh[(wave * 16 + e + 8 * hi) * KS + j * 16 + lrow] = tobf(s[j][e]);
    }

#pragma unroll
    for (int jd = 0; jd < 4; jd++)
#pragma unroll
      for (int e = 0; e < 8; e++) o[jd][e] *= sc[e];

    // O += P V
#pragma unroll
    for (int kk = 0; kk < 2; kk++) {
      const bf16* pr = &Psh[(wave * 16 + lrow) * KS + kk * 32];
      v16bf pf = cat16(*(const v8bf*)(pr + 8 * hi),
                       *(const v8bf*)(pr + 16 + 8 * hi));
#pragma unroll
      for (int jd = 0; jd < 4; jd++) {
        const int r = jd * 16 + lrow;  // lane's d column
        v16bf vf = cat16(*(const v8bf*)&VT[buf][r * KS + kk * 32 + 8 * hi],
                         *(const v8bf*)&VT[buf][r * KS + kk * 32 + 16 + 8 * hi]);
        o[jd] = __builtin_amdgcn_wmma_f32_16x16x32_bf16(
            false, pf, false, vf, (short)0, o[jd], false, false);
      }
    }
    __syncthreads();  // readers done before buf is DMA-overwritten
  }

#pragma unroll
  for (int e = 0; e < 8; e++) rl[e] = (rl[e] > 0.f) ? 1.0f / rl[e] : 0.f;
#pragma unroll
  for (int jd = 0; jd < 4; jd++)
#pragma unroll
    for (int e = 0; e < 8; e++) {
      const int t = q0 + e + 8 * hi;
      const int c = h * 64 + jd * 16 + lrow;
      Y[((size_t)b * T_ + t) * C_ + c] = tobf(o[jd][e] * rl[e]);
    }
}

// ---------------------------------------------------------------------------
// Host-side orchestration
// ---------------------------------------------------------------------------
extern "C" void kernel_launch(void* const* d_in, const int* in_sizes, int n_in,
                              void* d_out, int out_size, void* d_ws, size_t ws_size,
                              hipStream_t stream) {
  (void)in_sizes; (void)n_in; (void)out_size; (void)ws_size;

  const float* x      = (const float*)d_in[0];
  const unsigned char* pad = (const unsigned char*)d_in[1];
  const float* pocket = (const float*)d_in[2];
  const float* ln1_w  = (const float*)d_in[3];
  const float* ln1_b  = (const float*)d_in[4];
  const float* Wqkv   = (const float*)d_in[5];
  const float* bqkv   = (const float*)d_in[6];
  const float* Wo     = (const float*)d_in[7];
  const float* bo     = (const float*)d_in[8];
  const float* lnc_w  = (const float*)d_in[9];
  const float* lnc_b  = (const float*)d_in[10];
  const float* Wq     = (const float*)d_in[11];
  const float* bq     = (const float*)d_in[12];
  const float* Wkv    = (const float*)d_in[13];
  const float* bkv    = (const float*)d_in[14];
  const float* Wco    = (const float*)d_in[15];
  const float* bco    = (const float*)d_in[16];
  const float* ln2_w  = (const float*)d_in[17];
  const float* ln2_b  = (const float*)d_in[18];
  const float* Wfc    = (const float*)d_in[19];
  const float* bfc    = (const float*)d_in[20];
  const float* Wfp    = (const float*)d_in[21];
  const float* bfp    = (const float*)d_in[22];

  char* ws = (char*)d_ws;
  bf16* wqkv = (bf16*)(ws + 0);            // 6 MB  [3072,1024]
  bf16* wo   = (bf16*)(ws + 6291456);      // 2 MB  [1024,1024]
  bf16* wq   = (bf16*)(ws + 8388608);      // 2 MB
  bf16* wkv  = (bf16*)(ws + 10485760);     // 4 MB  [2048,1024]
  bf16* wco  = (bf16*)(ws + 14680064);     // 2 MB
  bf16* wfc  = (bf16*)(ws + 16777216);     // 8 MB  [4096,1024]
  bf16* wfp  = (bf16*)(ws + 25165824);     // 8 MB  [1024,4096]
  bf16* hln  = (bf16*)(ws + 33554432);     // 8 MB  (LN output, reused)
  bf16* qbuf = (bf16*)(ws + 41943040);     // 8 MB  q [B,H,T,D]
  bf16* kbuf = qbuf + 4194304;             // 8 MB  k [B,H,T,D]
  bf16* vbuf = kbuf + 4194304;             // 8 MB  v [B,H,D,T]
  bf16* pbf  = (bf16*)(ws + 67108864);     // 2 MB  pocket bf16
  bf16* ybuf = (bf16*)(ws + 69206016);     // 8 MB  attn out [B,T,C]
  bf16* hid  = (bf16*)(ws + 77594624);     // 32 MB MLP hidden
  unsigned char* pmask = (unsigned char*)(ws + 111149056);  // 1 KB

  // residual accumulator lives in d_out
  hipMemcpyAsync(d_out, x, (size_t)B_ * T_ * C_ * sizeof(float),
                 hipMemcpyDeviceToDevice, stream);

  // weight convert + transpose: fp32 [K,N] -> bf16 [N,K]
  cvt_t_kernel<<<dim3(96,  32), 256, 0, stream>>>(Wqkv, wqkv, 1024, 3072);
  cvt_t_kernel<<<dim3(32,  32), 256, 0, stream>>>(Wo,   wo,   1024, 1024);
  cvt_t_kernel<<<dim3(32,  32), 256, 0, stream>>>(Wq,   wq,   1024, 1024);
  cvt_t_kernel<<<dim3(64,  32), 256, 0, stream>>>(Wkv,  wkv,  1024, 2048);
  cvt_t_kernel<<<dim3(32,  32), 256, 0, stream>>>(Wco,  wco,  1024, 1024);
  cvt_t_kernel<<<dim3(128, 32), 256, 0, stream>>>(Wfc,  wfc,  1024, 4096);
  cvt_t_kernel<<<dim3(32, 128), 256, 0, stream>>>(Wfp,  wfp,  4096, 1024);

  pocket_kernel<<<B_ * TP_, 256, 0, stream>>>(pocket, pbf, pmask);

  // ---- self attention ----
  ln_kernel<<<B_ * T_, 256, 0, stream>>>((const float*)d_out, ln1_w, ln1_b, hln);
  gemm_bf16_wmma<<<dim3(24, 32), 256, 0, stream>>>(
      hln, wqkv, bqkv, (void*)qbuf, nullptr, 4096, 3072, 1024, 0, 10, 4194304, 2);
  attn_wmma<<<dim3(8, H_, B_), 256, 0, stream>>>(
      qbuf, kbuf, vbuf, pad, ybuf, T_, 1);
  gemm_bf16_wmma<<<dim3(8, 32), 256, 0, stream>>>(
      ybuf, wo, bo, d_out, pad, 4096, 1024, 1024, 1, 0, 0, -1);

  // ---- cross attention ----
  ln_kernel<<<B_ * T_, 256, 0, stream>>>((const float*)d_out, lnc_w, lnc_b, hln);
  gemm_bf16_wmma<<<dim3(8, 32), 256, 0, stream>>>(
      hln, wq, bq, (void*)qbuf, nullptr, 4096, 1024, 1024, 0, 10, 0, -1);
  gemm_bf16_wmma<<<dim3(16, 8), 256, 0, stream>>>(
      pbf, wkv, bkv, (void*)kbuf, nullptr, 1024, 2048, 1024, 0, 8, 1048576, 1);
  attn_wmma<<<dim3(8, H_, B_), 256, 0, stream>>>(
      qbuf, kbuf, kbuf + 1048576, pmask, ybuf, TP_, 0);
  gemm_bf16_wmma<<<dim3(8, 32), 256, 0, stream>>>(
      ybuf, wco, bco, d_out, pad, 4096, 1024, 1024, 1, 0, 0, -1);

  // ---- MLP ----
  ln_kernel<<<B_ * T_, 256, 0, stream>>>((const float*)d_out, ln2_w, ln2_b, hln);
  gemm_bf16_wmma<<<dim3(32, 32), 256, 0, stream>>>(
      hln, wfc, bfc, (void*)hid, nullptr, 4096, 4096, 1024, 2, 0, 0, -1);
  gemm_bf16_wmma<<<dim3(8, 32), 256, 0, stream>>>(
      hid, wfp, bfp, d_out, nullptr, 4096, 1024, 4096, 1, 0, 0, -1);
}
